// SimpleLSTM_79448305041737
// MI455X (gfx1250) — matrix operand order repaired and checked by reading, not compiled, not run on hardware
//
#include <hip/hip_runtime.h>
#include <hip/hip_bf16.h>

// ---------------------------------------------------------------------------
// Persistent WMMA LSTM for gfx1250 (MI455X), round 3.
//   grid = 16 WGs (one per 16-row batch tile), block = 512 (16 waves).
//   Wave jt owns hidden j-tile jt and ALL FOUR gates (n-tiles jt, 16+jt,
//   32+jt, 48+jt): C-fragment layout puts i,f,g,o for one (m,j) in the same
//   lane/slot -> fully in-register cell update, no cross-wave exchange.
//   Recurrent GEMM:  v_wmma_f32_16x16x32_bf16, B streamed from L2 via a
//   laundered ADDRESS_SPACE(1) pointer: loads stay in-loop (no LICM->spill,
//   round 1) AND encode as global_load_b128 (round 2 emitted flat_load_b128,
//   which couples LOADcnt with DScnt and orders against the LDS h loads).
//   Input GEMM (F=10->12): v_wmma_f32_16x16x4_f32, W_ih fragments resident.
//   h + X tiles double-buffered in LDS -> ONE barrier per step.
// ---------------------------------------------------------------------------

typedef __attribute__((ext_vector_type(16))) __bf16 v16bf;
typedef __attribute__((ext_vector_type(8)))  float  v8f;
typedef __attribute__((ext_vector_type(2)))  float  v2f;
typedef __attribute__((ext_vector_type(4)))  unsigned int v4u;

typedef const unsigned int __attribute__((address_space(1)))* guptr;
typedef const v4u          __attribute__((address_space(1)))* gv4uptr;

union Frag16 { v16bf v; v4u q[2]; };

#define B_   256
#define T_   512
#define F_   10
#define FP_  12          // F padded to multiple of 4 for 16x16x4 f32 WMMA
#define H_   256
#define P_   264         // h row pitch (bf16 elems): 256 + 8 pad for LDS banks
#define HSZ  (16 * P_)
#define WHH_FRAG_DW (64 * 8 * 32 * 8)          // 131072 dwords (512 KB)
#define WIH_FRAG_DW (64 * 3 * 32 * 2)          // 12288 dwords  (48 KB)

__device__ __forceinline__ unsigned short f2bf(float f) {
  __bf16 h = (__bf16)f;
  return __builtin_bit_cast(unsigned short, h);
}
__device__ __forceinline__ float bf2f(unsigned short u) {
  unsigned int x = ((unsigned int)u) << 16;
  return __builtin_bit_cast(float, x);
}
__device__ __forceinline__ float sigmoid_f(float x) {
  x = fminf(fmaxf(x, -30.f), 30.f);
  return __builtin_amdgcn_rcpf(1.f + __expf(-x));
}
__device__ __forceinline__ float tanh_f(float x) {
  x = fminf(fmaxf(x, -15.f), 15.f);
  float e = __expf(-2.f * x);
  return (1.f - e) * __builtin_amdgcn_rcpf(1.f + e);
}

// --- prep: pack W_hh into bf16 B-fragments, W_ih into f32 B-fragments ------
// bf16 B (32x16) frag (nt,kc): lane l -> n = nt*16+(l&15); dword v ->
//   k = kc*32 + (l>=16 ? 16:0) + 2v ("row striped across lanes in one VGPR").
// f32 B (4x16) frag (nt,kc3): lane l -> n = nt*16+(l&15); dword v(0,1) ->
//   k = kc3*4 + (l>=16 ? 2:0) + v, zero-padded for k >= 10.
__global__ void pack_weights_kernel(const float* __restrict__ Whh,
                                    const float* __restrict__ Wih,
                                    unsigned int* __restrict__ ws) {
  int idx = blockIdx.x * blockDim.x + threadIdx.x;
  if (idx < WHH_FRAG_DW) {
    int v    = idx & 7;
    int lane = (idx >> 3) & 31;
    int kc   = (idx >> 8) & 7;
    int nt   = idx >> 11;
    int n = nt * 16 + (lane & 15);
    int k = kc * 32 + ((lane >> 4) ? 16 : 0) + 2 * v;
    unsigned int lo = f2bf(Whh[n * H_ + k]);
    unsigned int hi = f2bf(Whh[n * H_ + k + 1]);
    ws[idx] = lo | (hi << 16);
  } else if (idx < WHH_FRAG_DW + WIH_FRAG_DW) {
    int j = idx - WHH_FRAG_DW;          // ((nt*3+kc3)*32 + lane)*2 + v
    int v    = j & 1;
    int lane = (j >> 1) & 31;
    int kc3  = (j >> 6) % 3;
    int nt   = j / 192;
    int n = nt * 16 + (lane & 15);
    int k = kc3 * 4 + ((lane >> 4) ? 2 : 0) + v;
    float val = (k < F_) ? Wih[n * F_ + k] : 0.f;
    ws[idx] = __builtin_bit_cast(unsigned int, val);
  }
}

__global__ __launch_bounds__(512, 1)
void lstm_persistent_kernel(const float* __restrict__ X,
                            const float* __restrict__ b_ih,
                            const float* __restrict__ b_hh,
                            const float* __restrict__ Wd,
                            const float* __restrict__ bd,
                            const unsigned int* __restrict__ wsFrag,
                            float* __restrict__ out) {
  __shared__ __align__(16) unsigned short h_lds[2 * HSZ];  // ping-pong h (bf16)
  __shared__ __align__(16) float s_x[2 * 16 * FP_];        // ping-pong X tile

  const int tid  = threadIdx.x;
  const int wave = tid >> 5;         // 0..15 == hidden j-tile jt
  const int lane = tid & 31;
  const int half = lane >> 4;
  const int li   = lane & 15;
  const int jt   = wave;
  const int bbase = blockIdx.x * 16;

  // Resident per-lane constants: W_ih f32 B-fragments (24 VGPRs) + fused bias.
  v2f   wihf[4][3];
  float bias[4];
#pragma unroll
  for (int q = 0; q < 4; ++q) {
    int n = q * H_ + jt * 16 + li;
    bias[q] = b_ih[n] + b_hh[n];
    int nt = q * 16 + jt;
    const unsigned int* wp = wsFrag + WHH_FRAG_DW;
#pragma unroll
    for (int kc3 = 0; kc3 < 3; ++kc3) {
      const unsigned int* p = wp + (((size_t)nt * 3 + kc3) * 32 + lane) * 2;
      wihf[q][kc3][0] = __builtin_bit_cast(float, p[0]);
      wihf[q][kc3][1] = __builtin_bit_cast(float, p[1]);
    }
  }

  // Prologue: h0 = 0 (buffer 0), zero X pads, stage X[:,0,:].
  for (int i = tid; i < HSZ; i += 512) h_lds[i] = 0;
  if (tid < 2 * 16 * FP_) s_x[tid] = 0.f;
  __syncthreads();
  if (tid < 16 * F_) {
    int m = tid / F_, f = tid - m * F_;
    s_x[m * FP_ + f] = X[((size_t)(bbase + m) * T_ + 0) * F_ + f];
  }

  float c[8];
#pragma unroll
  for (int r = 0; r < 8; ++r) c[r] = 0.f;

  // Laundered base (SGPR): stops LICM from hoisting the per-step W_hh
  // fragment loads (round 1: hoist -> spill -> per-step scratch reloads).
  // Re-cast to address_space(1) each step so they encode as global_load,
  // not flat_load (round 2: flat couples LOADcnt+DScnt with the LDS path).
  uintptr_t fb = (uintptr_t)wsFrag;

#pragma unroll 2
  for (int t = 0; t < T_; ++t) {
    __syncthreads();   // h[t-1] (read buf) + X[t] (cur buf) ready
    const int cur = t & 1, nxt = cur ^ 1;
    const unsigned short* hread  = h_lds + cur * HSZ;
    unsigned short*       hwrite = h_lds + nxt * HSZ;

    // Stage X[:, t+1, :] into the other buffer (its own prefetch).
    if (t + 1 < T_ && tid < 16 * F_) {
      int m = tid / F_, f = tid - m * F_;
      s_x[nxt * 16 * FP_ + m * FP_ + f] =
          X[((size_t)(bbase + m) * T_ + (t + 1)) * F_ + f];
    }

    // Pipelined output dense for step t-1 (h[t-1] lives in hread).
    if (t > 0 && tid < 48) {
      int m = tid / 3, o = tid - m * 3;
      float s = bd[o];
#pragma unroll 8
      for (int j = 0; j < H_; ++j)
        s += bf2f(hread[m * P_ + j]) * Wd[o * H_ + j];
      out[((size_t)(bbase + m) * T_ + (t - 1)) * 3 + o] = s;
    }

    asm volatile("" : "+s"(fb));
    guptr fp = (guptr)fb;

    v8f acc[4] = {v8f{}, v8f{}, v8f{}, v8f{}};

    // Input GEMM: xg += X_t[16x12] * W_ih^T, exact f32 WMMA (K=4 x 3).
#pragma unroll
    for (int kc3 = 0; kc3 < 3; ++kc3) {
      v2f ax = *(const v2f*)&s_x[cur * 16 * FP_ + li * FP_ + kc3 * 4 + half * 2];
#pragma unroll
      for (int q = 0; q < 4; ++q)
        acc[q] = __builtin_amdgcn_wmma_f32_16x16x4_f32(
            false, ax, false, wihf[q][kc3], (short)0, acc[q], false, false);
    }

    // Recurrent GEMM: gates += h[16x256] * W_hh^T, bf16 WMMA (K=32 x 8),
    // B-fragments streamed from L2 each step (loop-invariant, off the
    // critical h-dependency path; global encoding -> LOADcnt only).
#pragma unroll
    for (int kc = 0; kc < 8; ++kc) {
      Frag16 a;
      int kb = kc * 32 + half * 16;
      a.q[0] = *(const v4u*)&hread[li * P_ + kb];
      a.q[1] = *(const v4u*)&hread[li * P_ + kb + 8];
#pragma unroll
      for (int q = 0; q < 4; ++q) {
        Frag16 b;
        guptr bp = fp + (((size_t)(q * 16 + jt) * 8 + kc) * 32 + lane) * 8;
        b.q[0] = *(gv4uptr)(bp);
        b.q[1] = *(gv4uptr)(bp + 4);
        acc[q] = __builtin_amdgcn_wmma_f32_16x16x32_bf16(
            false, a.v, false, b.v, (short)0, acc[q], false, false);
      }
    }

    // Cell update, fully in-lane: i,f,g,o for (m = r+8*half, j = jt*16+li)
    // all sit in slot r of acc[0..3].
#pragma unroll
    for (int r = 0; r < 8; ++r) {
      float gi = sigmoid_f(acc[0][r] + bias[0]);
      float gf = sigmoid_f(acc[1][r] + bias[1]);
      float gg = tanh_f  (acc[2][r] + bias[2]);
      float go = sigmoid_f(acc[3][r] + bias[3]);
      c[r] = fmaf(gf, c[r], gi * gg);
      float hv = go * tanh_f(c[r]);
      hwrite[(r + half * 8) * P_ + jt * 16 + li] = f2bf(hv);
    }
  }

  // Final output row (t = T-1; h[T-1] is in buffer T_&1 == 0).
  __syncthreads();
  if (tid < 48) {
    const unsigned short* hfin = h_lds + (T_ & 1) * HSZ;
    int m = tid / 3, o = tid - m * 3;
    float s = bd[o];
#pragma unroll 8
    for (int j = 0; j < H_; ++j)
      s += bf2f(hfin[m * P_ + j]) * Wd[o * H_ + j];
    out[((size_t)(bbase + m) * T_ + (T_ - 1)) * 3 + o] = s;
  }
}

extern "C" void kernel_launch(void* const* d_in, const int* in_sizes, int n_in,
                              void* d_out, int out_size, void* d_ws, size_t ws_size,
                              hipStream_t stream) {
  const float* X     = (const float*)d_in[0];
  const float* W_ih  = (const float*)d_in[1];
  const float* W_hh  = (const float*)d_in[2];
  const float* b_ih  = (const float*)d_in[3];
  const float* b_hh  = (const float*)d_in[4];
  const float* Wd    = (const float*)d_in[5];
  const float* bd    = (const float*)d_in[6];
  float* out = (float*)d_out;
  unsigned int* frag = (unsigned int*)d_ws;   // 512 KB W_hh bf16 + 48 KB W_ih f32

  const int prepN = WHH_FRAG_DW + WIH_FRAG_DW;
  pack_weights_kernel<<<dim3((prepN + 255) / 256), dim3(256), 0, stream>>>(
      W_hh, W_ih, frag);
  lstm_persistent_kernel<<<dim3(B_ / 16), dim3(512), 0, stream>>>(
      X, b_ih, b_hh, Wd, bd, frag, out);
}
